// DistanceClusterLoss_2851858284970
// MI455X (gfx1250) — matrix operand order repaired and checked by reading, not compiled, not run on hardware
//
#include <hip/hip_runtime.h>
#include <stdint.h>

// ---------------------------------------------------------------------------
// DistanceClusterLoss on MI455X (gfx1250, wave32, WMMA + TDM)
//
// out = (1/N) * sum_{i!=j} sign(l_i,l_j) * ||x_i - x_j||,  x:(8192,200) f32.
// Gram matrix via V_WMMA_F32_16X16X32_F16 with f32 = hi+lo f16 split
// (3 WMMAs per K=32 step, lo*lo dropped -> ~2^-22 rel err).
// B panel (shared by all 8 waves of a block) is staged into LDS by the
// Tensor Data Mover (tensor_load_to_lds), double buffered, synchronized
// with s_wait_tensorcnt + workgroup barriers. A tiles stream from L2 with
// hoisted base pointers + immediate offsets.
// ---------------------------------------------------------------------------

typedef _Float16 v16h __attribute__((ext_vector_type(16)));
typedef _Float16 v8h  __attribute__((ext_vector_type(8)));
typedef float    v8f  __attribute__((ext_vector_type(8)));
typedef unsigned int u32x4 __attribute__((ext_vector_type(4)));
typedef int          i32x4 __attribute__((ext_vector_type(4)));
typedef int          i32x8 __attribute__((ext_vector_type(8)));

#define N_ROWS 8192
#define D_IN   200
#define KP     224            // padded K: 7 * 32
#define TBLK   128            // 128x128 output tile per workgroup
#define NTILE  (N_ROWS / TBLK)
#define BROW   40             // LDS B-panel row stride in f16 (80B, bank-spread)

#define HAVE_TDM __has_builtin(__builtin_amdgcn_tensor_load_to_lds)

// ---------------------------------------------------------------------------
// Kernel 1: f32 -> f16 hi/lo split (K padded to 224) + row squared norms.
// One wave per row.
// ---------------------------------------------------------------------------
__global__ void __launch_bounds__(256) pack_kernel(
    const float* __restrict__ preds,
    _Float16* __restrict__ xhi,
    _Float16* __restrict__ xlo,
    float* __restrict__ sq)
{
    const int wave = threadIdx.x >> 5;
    const int lane = threadIdx.x & 31;
    const int row  = blockIdx.x * 8 + wave;
    const float* src = preds + (size_t)row * D_IN;

    float ssq = 0.f;
#pragma unroll
    for (int t = 0; t < 7; ++t) {
        const int k = lane + t * 32;
        const float x = (k < D_IN) ? src[k] : 0.f;
        ssq += x * x;
        const _Float16 h = (_Float16)x;
        const _Float16 l = (_Float16)(x - (float)h);
        xhi[(size_t)row * KP + k] = h;
        xlo[(size_t)row * KP + k] = l;
    }
#pragma unroll
    for (int off = 16; off > 0; off >>= 1)
        ssq += __shfl_xor(ssq, off, 32);
    if (lane == 0) sq[row] = ssq;
}

// ---------------------------------------------------------------------------
// TDM helpers
// ---------------------------------------------------------------------------
__device__ __forceinline__ unsigned lds_off(const void* p)
{
    // generic pointer to LDS: high 32 bits = shared aperture, low 32 = offset
    return (unsigned)(unsigned long long)p;
}

#if HAVE_TDM
__device__ __forceinline__ void tdm_issue(unsigned ldsAddr,
                                          const _Float16* src,
                                          i32x8 g1)
{
    const unsigned long long ga = (unsigned long long)src;
    u32x4 g0;
    g0[0] = 1u;                                   // count=1, no gather
    g0[1] = ldsAddr;                              // lds_addr  [63:32]
    g0[2] = (unsigned)ga;                         // global_addr[31:0]
    g0[3] = (unsigned)((ga >> 32) & 0x01ffffffULL)// global_addr[56:32]
          | 0x80000000u;                          // type=2 at [127:126]
    i32x4 z4 = {0, 0, 0, 0};
    i32x8 z8 = {0, 0, 0, 0, 0, 0, 0, 0};
    __builtin_amdgcn_tensor_load_to_lds(g0, g1, z4, z4, z8, 0);
}
#endif

// ---------------------------------------------------------------------------
// Kernel 2: 128x128 tile of the pairwise sum.
// 8 waves: wave w -> (wi = w&3) 32-row slab, (wj = w>>2) 64-col slab.
// B panel (128 cols x 32 k, hi+lo) staged in LDS by TDM, double buffered.
// ---------------------------------------------------------------------------
__global__ void __launch_bounds__(256) gram_tile_kernel(
    const _Float16* __restrict__ xhi,
    const _Float16* __restrict__ xlo,
    const float* __restrict__ sq,
    const long long* __restrict__ labels,
    float* __restrict__ partials)
{
    __shared__ _Float16 bufH[2][TBLK * BROW];     // 2 x 10KB
    __shared__ _Float16 bufL[2][TBLK * BROW];
    __shared__ float sWave[8];

    const int bi   = blockIdx.x;
    const int bj   = blockIdx.y;
    const int bIdx = bj * gridDim.x + bi;

    if (bj < bi) {                                // mirror handled by weight 2
        if (threadIdx.x == 0) partials[bIdx] = 0.f;
        return;
    }

    const int wave = threadIdx.x >> 5;
    const int lane = threadIdx.x & 31;
    const int wi   = wave & 3;
    const int wj   = wave >> 2;
    const int i0   = bi * TBLK + wi * 32;
    const int j0   = bj * TBLK + wj * 64;
    const int g    = lane >> 4;
    const int r    = lane & 15;

    // ---- hoisted A base pointers (k-invariant; k-loop uses imm offsets)
    const _Float16* pAh[2];
    const _Float16* pAl[2];
#pragma unroll
    for (int si = 0; si < 2; ++si) {
        const int off = (i0 + 16 * si + r) * KP + 8 * g;
        pAh[si] = xhi + off;
        pAl[si] = xlo + off;
    }
    // ---- hoisted LDS element offsets for B reads
    int colB[4];
#pragma unroll
    for (int sj = 0; sj < 4; ++sj)
        colB[sj] = (wj * 64 + 16 * sj + r) * BROW + 8 * g;

    const _Float16* gBh = xhi + (size_t)(bj * TBLK) * KP;   // B panel base
    const _Float16* gBl = xlo + (size_t)(bj * TBLK) * KP;

#if HAVE_TDM
    // D# group1: data_size=2B, pad 4 DWORDs after every 16 DWORDs (64B row ->
    // 80B LDS stride), tensor (224 x 8192) stride 224, tile 32 x 128.
    i32x8 g1;
    g1[0] = (1 << 16) | (1 << 20) | (3 << 22) | (3 << 25);
    g1[1] = (KP & 0xffff) << 16;        // tensor_dim0[15:0] @ bits 63:48
    g1[2] = (N_ROWS & 0xffff) << 16;    // tensor_dim1[15:0] @ bits 95:80
    g1[3] = (32 << 16);                 // tile_dim0 = 32    @ bits 127:112
    g1[4] = TBLK;                       // tile_dim1 = 128   @ bits 143:128
    g1[5] = KP;                         // tensor_dim0_stride[31:0]
    g1[6] = 0;
    g1[7] = 0;

    if (wave == 0) {                    // prologue: fill buffer 0 (k0 = 0)
        tdm_issue(lds_off(&bufH[0][0]), gBh, g1);
        tdm_issue(lds_off(&bufL[0][0]), gBl, g1);
    }
#else
    // fallback prologue: cooperative fill of buffer 0 (k0 = 0)
    {
        const int crow  = threadIdx.x >> 1;           // 0..127
        const int chalf = threadIdx.x & 1;            // 16-element half
        const int gofs  = crow * KP + 16 * chalf;
        const int lofs  = crow * BROW + 16 * chalf;
        *(v8h*)(&bufH[0][lofs])     = *(const v8h*)(gBh + gofs);
        *(v8h*)(&bufH[0][lofs + 8]) = *(const v8h*)(gBh + gofs + 8);
        *(v8h*)(&bufL[0][lofs])     = *(const v8h*)(gBl + gofs);
        *(v8h*)(&bufL[0][lofs + 8]) = *(const v8h*)(gBl + gofs + 8);
    }
#endif

    v8f acc[2][4];
#pragma unroll
    for (int si = 0; si < 2; ++si)
#pragma unroll
        for (int sj = 0; sj < 4; ++sj)
#pragma unroll
            for (int e = 0; e < 8; ++e) acc[si][sj][e] = 0.f;

#pragma unroll
    for (int ks = 0; ks < 7; ++ks) {
        const int k0  = ks * 32;
        const int cur = ks & 1;
        const int nxt = cur ^ 1;

#if HAVE_TDM
        if (wave == 0) {
            if (ks + 1 < 7) {           // prefetch next panel into other buffer
                tdm_issue(lds_off(&bufH[nxt][0]), gBh + (k0 + 32), g1);
                tdm_issue(lds_off(&bufL[nxt][0]), gBl + (k0 + 32), g1);
                __builtin_amdgcn_s_wait_tensorcnt(2);   // current panel done
            } else {
                __builtin_amdgcn_s_wait_tensorcnt(0);   // last panel done
            }
        }
        __syncthreads();                // buf[cur] ready for all waves
#else
        __syncthreads();                // buf[cur] ready (stored last iter)
#endif

        // ---- A tiles: 16-bit A layout, imm-offset loads from L2
        v16h aH[2], aL[2];
#pragma unroll
        for (int si = 0; si < 2; ++si) {
            const v8h h0 = *(const v8h*)(pAh[si] + k0);
            const v8h h1 = *(const v8h*)(pAh[si] + k0 + 16);
            aH[si] = __builtin_shufflevector(h0, h1, 0,1,2,3,4,5,6,7,
                                                     8,9,10,11,12,13,14,15);
            const v8h l0 = *(const v8h*)(pAl[si] + k0);
            const v8h l1 = *(const v8h*)(pAl[si] + k0 + 16);
            aL[si] = __builtin_shufflevector(l0, l1, 0,1,2,3,4,5,6,7,
                                                     8,9,10,11,12,13,14,15);
        }

        // ---- B tiles from LDS panel; 24 WMMAs
#pragma unroll
        for (int sj = 0; sj < 4; ++sj) {
            const _Float16* bh = &bufH[cur][colB[sj]];
            const _Float16* bl = &bufL[cur][colB[sj]];
            const v16h bH = __builtin_shufflevector(
                *(const v8h*)bh, *(const v8h*)(bh + 16),
                0,1,2,3,4,5,6,7,8,9,10,11,12,13,14,15);
            const v16h bL = __builtin_shufflevector(
                *(const v8h*)bl, *(const v8h*)(bl + 16),
                0,1,2,3,4,5,6,7,8,9,10,11,12,13,14,15);
#pragma unroll
            for (int si = 0; si < 2; ++si) {
                acc[si][sj] = __builtin_amdgcn_wmma_f32_16x16x32_f16(
                    false, aH[si], false, bH, (short)0, acc[si][sj], false, false);
                acc[si][sj] = __builtin_amdgcn_wmma_f32_16x16x32_f16(
                    false, aH[si], false, bL, (short)0, acc[si][sj], false, false);
                acc[si][sj] = __builtin_amdgcn_wmma_f32_16x16x32_f16(
                    false, aL[si], false, bH, (short)0, acc[si][sj], false, false);
            }
        }

#if !HAVE_TDM
        // cooperative fallback: copy next panel (hi+lo) into buf[nxt]
        if (ks + 1 < 7) {
            const int crow  = threadIdx.x >> 1;       // 0..127
            const int chalf = threadIdx.x & 1;        // 16-element half
            const int gofs  = crow * KP + (k0 + 32) + 16 * chalf;
            const int lofs  = crow * BROW + 16 * chalf;
            *(v8h*)(&bufH[nxt][lofs])     = *(const v8h*)(gBh + gofs);
            *(v8h*)(&bufH[nxt][lofs + 8]) = *(const v8h*)(gBh + gofs + 8);
            *(v8h*)(&bufL[nxt][lofs])     = *(const v8h*)(gBl + gofs);
            *(v8h*)(&bufL[nxt][lofs + 8]) = *(const v8h*)(gBl + gofs + 8);
        }
#endif
        __syncthreads();                // all reads of buf[cur] done
    }

    // ---- Epilogue: C layout VGPR e, lane -> M = e + 8*g, N = r.
    float part = 0.f;
    const float wFactor = (bi == bj) ? 1.f : 2.f;

    float     sqj[4];
    long long labj[4];
#pragma unroll
    for (int sj = 0; sj < 4; ++sj) {
        const int j = j0 + 16 * sj + r;
        sqj[sj]  = sq[j];
        labj[sj] = labels[j];
    }

#pragma unroll
    for (int si = 0; si < 2; ++si) {
#pragma unroll
        for (int e = 0; e < 8; ++e) {
            const int i = i0 + 16 * si + e + 8 * g;
            const float     sqi = sq[i];
            const long long li  = labels[i];
#pragma unroll
            for (int sj = 0; sj < 4; ++sj) {
                const int j = j0 + 16 * sj + r;
                const float d2   = sqi + sqj[sj] - 2.0f * acc[si][sj][e];
                const float dist = sqrtf(fmaxf(d2, 0.f));
                const float s    = (li == labj[sj]) ? 1.f : -1.f;
                const float w    = (i == j) ? 0.f : wFactor;
                part += w * s * dist;
            }
        }
    }

#pragma unroll
    for (int off = 16; off > 0; off >>= 1)
        part += __shfl_xor(part, off, 32);

    if (lane == 0) sWave[wave] = part;
    __syncthreads();
    if (threadIdx.x == 0) {
        float t = 0.f;
#pragma unroll
        for (int w = 0; w < 8; ++w) t += sWave[w];
        partials[bIdx] = t;
    }
}

// ---------------------------------------------------------------------------
// Kernel 3: deterministic tree reduction of the block partials.
// ---------------------------------------------------------------------------
__global__ void __launch_bounds__(256) reduce_kernel(
    const float* __restrict__ partials, int n, float* __restrict__ out)
{
    __shared__ float smem[256];
    float s = 0.f;
    for (int t = threadIdx.x; t < n; t += 256) s += partials[t];
    smem[threadIdx.x] = s;
    __syncthreads();
#pragma unroll
    for (int w = 128; w > 0; w >>= 1) {
        if ((int)threadIdx.x < w) smem[threadIdx.x] += smem[threadIdx.x + w];
        __syncthreads();
    }
    if (threadIdx.x == 0) out[0] = smem[0] / (float)N_ROWS;
}

// ---------------------------------------------------------------------------
extern "C" void kernel_launch(void* const* d_in, const int* in_sizes, int n_in,
                              void* d_out, int out_size, void* d_ws, size_t ws_size,
                              hipStream_t stream)
{
    const float*     preds  = (const float*)d_in[0];
    const long long* labels = (const long long*)d_in[1];
    float* out = (float*)d_out;

    char* ws = (char*)d_ws;
    const size_t xBytes = (size_t)N_ROWS * KP * sizeof(_Float16);
    _Float16* xhi      = (_Float16*)(ws);
    _Float16* xlo      = (_Float16*)(ws + xBytes);
    float*    sq       = (float*)   (ws + 2 * xBytes);
    float*    partials = (float*)   (ws + 2 * xBytes + N_ROWS * sizeof(float));

    pack_kernel<<<N_ROWS / 8, 256, 0, stream>>>(preds, xhi, xlo, sq);

    dim3 grid(NTILE, NTILE);
    gram_tile_kernel<<<grid, 256, 0, stream>>>(xhi, xlo, sq, labels, partials);

    reduce_kernel<<<1, 256, 0, stream>>>(partials, NTILE * NTILE, out);
}
